// EdgePathNN_66030827209109
// MI455X (gfx1250) — compile-verified, matrix operand order checked by reading, not run on hardware
//
#include <hip/hip_runtime.h>

#define Hdim 64
#define Nn   50000
#define Pp   200000
#define Bb   256
#define NCc  10
#define INVSTD 0.9999950000374997f  /* 1/sqrt(1+1e-5) */

typedef __attribute__((ext_vector_type(16))) _Float16 v16h;
typedef __attribute__((ext_vector_type(8)))  _Float16 v8h;
typedef __attribute__((ext_vector_type(4)))  _Float16 v4h;
typedef __attribute__((ext_vector_type(8)))  float    v8f;

// Load a 16x32 f16 WMMA fragment (A or B) from a row-major buffer.
// Lane layout per CDNA5 ISA 7.12.2: lanes 0-15 hold K {0..7,16..23}, lanes 16-31
// hold K {8..15,24..31}, two halves per VGPR.
__device__ __forceinline__ v16h load_frag(const _Float16* base, int row0, int ld,
                                          int k0, int lane) {
  const int n15 = lane & 15;
  const int hi  = (lane >> 4) & 1;
  const _Float16* p = base + (size_t)(row0 + n15) * ld + k0 + hi * 8;
  union { v16h v; v8h h[2]; } u;
  u.h[0] = *(const v8h*)(p);
  u.h[1] = *(const v8h*)(p + 16);
  return u.v;
}

// Fast activations: v_exp_f32 + v_rcp_f32 instead of libm tanhf / precise div.
__device__ __forceinline__ float fast_rcp(float x) { return __builtin_amdgcn_rcpf(x); }
__device__ __forceinline__ float sigm(float x) {
  return fast_rcp(1.0f + __expf(-x));
}
__device__ __forceinline__ float fast_tanh(float x) {
  float e = __expf(-2.0f * x);
  return (1.0f - e) * fast_rcp(1.0f + e);
}

// ---------------------------------------------------------------------------
// prep: convert weights to f16 (Wih ++ Whh contiguous), combined LSTM bias
// ---------------------------------------------------------------------------
__global__ void prep_kernel(const float* __restrict__ Wih, const float* __restrict__ Whh,
                            const float* __restrict__ mW1, const float* __restrict__ mW2,
                            const float* __restrict__ bih, const float* __restrict__ bhh,
                            _Float16* __restrict__ Wih_h, _Float16* __restrict__ Whh_h,
                            _Float16* __restrict__ W1_h, _Float16* __restrict__ W2_h,
                            float* __restrict__ bsum) {
  int i = blockIdx.x * blockDim.x + threadIdx.x;
  if (i < 256 * 128) Wih_h[i] = (_Float16)Wih[i];
  if (i < 256 * 64)  Whh_h[i] = (_Float16)Whh[i];
  if (i < 3 * 64 * 64) { W1_h[i] = (_Float16)mW1[i]; W2_h[i] = (_Float16)mW2[i]; }
  if (i < 256) bsum[i] = bih[i] + bhh[i];
}

// ---------------------------------------------------------------------------
// node embedding gather: W0[n,h] = feat_emb[x_ids[n], h]
// ---------------------------------------------------------------------------
__global__ void embed_kernel(const int* __restrict__ x_ids,
                             const float* __restrict__ feat_emb,
                             float* __restrict__ W0) {
  int i = blockIdx.x * blockDim.x + threadIdx.x;
  if (i < Nn * Hdim) W0[i] = feat_emb[x_ids[i >> 6] * Hdim + (i & 63)];
}

// ---------------------------------------------------------------------------
// W_in for layer: layer 0 -> W0 ; else 0.5*W0 - 0.5*sum(prev conv results)
// ---------------------------------------------------------------------------
__global__ void win_kernel(const float* __restrict__ W0, const float* __restrict__ convr,
                           float* __restrict__ Win, int layer) {
  int i = blockIdx.x * blockDim.x + threadIdx.x;
  if (i >= Nn * Hdim) return;
  if (layer == 0) { Win[i] = W0[i]; return; }
  float s = convr[i];
  if (layer == 2) s += convr[(size_t)Nn * Hdim + i];
  Win[i] = 0.5f * (W0[i] - s);
}

// ---------------------------------------------------------------------------
// LSTM over paths, WMMA gate GEMMs, scatter-add of final h into agg.
// One wave handles 16 paths; block = 8 waves = 128 paths.
// Weights live in LDS (one shared object with the tiles so the weight ds_loads
// cannot be hoisted out of the t-loop and spilled).
// ---------------------------------------------------------------------------
__global__ __launch_bounds__(256) void lstm_kernel(
    const float* __restrict__ Win, const float* __restrict__ bond_emb,
    const int* __restrict__ edge_attr_ids,
    const int* __restrict__ paths, const int* __restrict__ eidx,
    const _Float16* __restrict__ Wg,   // Wih (256x128) ++ Whh (256x64), f16
    const float* __restrict__ bsum, float* __restrict__ agg, int L, int P) {
  // [0,32768): Wih  [32768,49152): Whh  [49152,...): per-wave x(2048)+h(1024)
  __shared__ _Float16 smem[49152 + 8 * 3072];  // 144 KB
  const int tid  = threadIdx.x;
  const int wave = tid >> 5;
  const int lane = tid & 31;
  const int n15  = lane & 15;
  const int hi   = lane >> 4;
  const int p0   = (blockIdx.x * 8 + wave) * 16;

  // cooperative copy of weights into LDS (49152 halves = 6144 x b128)
  {
    v8h* dst = (v8h*)smem;
    const v8h* src = (const v8h*)Wg;
    for (int i = tid; i < 6144; i += 256) dst[i] = src[i];
  }
  _Float16* xT = smem + 49152 + wave * 3072;
  _Float16* hT = xT + 2048;
  const _Float16* WihL = smem;
  const _Float16* WhhL = smem + 32768;

  for (int i = lane; i < 16 * 64; i += 32) hT[i] = (_Float16)0.f;
  __syncthreads();

  float cst[4][8];
#pragma unroll
  for (int q = 0; q < 4; ++q)
#pragma unroll
    for (int r = 0; r < 8; ++r) cst[q][r] = 0.f;

  for (int t = 0; t < L; ++t) {
    // ---- stage x_cat = [W_in[path[t]] | edge_emb] as f16 in LDS ----
    {
      int m = n15;
      int p = p0 + m; if (p >= P) p = P - 1;
      if (hi == 0) {
        int node = paths[p * L + t];
        const float4* src4 = (const float4*)(Win + (size_t)node * Hdim);
#pragma unroll
        for (int c = 0; c < 16; ++c) {
          float4 v = src4[c];
          v4h hv; hv.x = (_Float16)v.x; hv.y = (_Float16)v.y;
          hv.z = (_Float16)v.z; hv.w = (_Float16)v.w;
          *(v4h*)(xT + m * 128 + c * 4) = hv;
        }
      } else {
        if (t == 0) {
          v4h z; z.x = z.y = z.z = z.w = (_Float16)0.f;
#pragma unroll
          for (int c = 0; c < 16; ++c) *(v4h*)(xT + m * 128 + 64 + c * 4) = z;
        } else {
          int e   = eidx[p * (L - 1) + (t - 1)];
          int bid = edge_attr_ids[e];
          const float4* src4 = (const float4*)(bond_emb + (size_t)bid * Hdim);
#pragma unroll
          for (int c = 0; c < 16; ++c) {
            float4 v = src4[c];
            v4h hv; hv.x = (_Float16)v.x; hv.y = (_Float16)v.y;
            hv.z = (_Float16)v.z; hv.w = (_Float16)v.w;
            *(v4h*)(xT + m * 128 + 64 + c * 4) = hv;
          }
        }
      }
    }

    // ---- A fragments: x_cat (K=128 -> 4 chunks), h (K=64 -> 2 chunks) ----
    v16h aX[4];
#pragma unroll
    for (int k = 0; k < 4; ++k) aX[k] = load_frag(xT, 0, 128, k * 32, lane);
    v16h aH[2];
#pragma unroll
    for (int k = 0; k < 2; ++k) aH[k] = load_frag(hT, 0, 64, k * 32, lane);

    // ---- gates: for each hidden 16-col chunk q, compute i,f,g,o tiles ----
#pragma unroll
    for (int q = 0; q < 4; ++q) {
      v8f acc[4];
#pragma unroll
      for (int G = 0; G < 4; ++G) {
        const int nt = G * 4 + q;            // gate-major n-tile in [0,16)
        float bz = bsum[nt * 16 + n15];
        v8f a;
#pragma unroll
        for (int r = 0; r < 8; ++r) a[r] = bz;
#pragma unroll
        for (int k = 0; k < 4; ++k) {
          v16h b = load_frag(WihL, nt * 16, 128, k * 32, lane);
          a = __builtin_amdgcn_wmma_f32_16x16x32_f16(false, aX[k], false, b,
                                                     (short)0, a, false, false);
        }
#pragma unroll
        for (int k = 0; k < 2; ++k) {
          v16h b = load_frag(WhhL, nt * 16, 64, k * 32, lane);
          a = __builtin_amdgcn_wmma_f32_16x16x32_f16(false, aH[k], false, b,
                                                     (short)0, a, false, false);
        }
        acc[G] = a;
      }
#pragma unroll
      for (int r = 0; r < 8; ++r) {
        float ig = sigm(acc[0][r]);
        float fg = sigm(acc[1][r]);
        float gg = fast_tanh(acc[2][r]);
        float og = sigm(acc[3][r]);
        float c  = fg * cst[q][r] + ig * gg;
        cst[q][r] = c;
        hT[(hi * 8 + r) * 64 + q * 16 + n15] = (_Float16)(og * fast_tanh(c));
      }
    }
  }

  // ---- segment_sum: atomic scatter-add of final h (in LDS) into agg ----
#pragma unroll
  for (int r = 0; r < 8; ++r) {
    int m = hi * 8 + r;
    int p = p0 + m;
    if (p < P) {
      int dest = paths[p * L + (L - 1)];
#pragma unroll
      for (int q = 0; q < 4; ++q) {
        float v = (float)hT[m * 64 + q * 16 + n15];
        atomicAdd(&agg[(size_t)dest * Hdim + q * 16 + n15], v);
      }
    }
  }
}

// ---------------------------------------------------------------------------
// per-node MLP: bn(agg) -> W1 -> bn1 -> relu -> W2 -> bn2 -> relu (WMMA)
// ---------------------------------------------------------------------------
__global__ __launch_bounds__(256) void mlp_kernel(
    const float* __restrict__ agg,
    const float* __restrict__ cbg, const float* __restrict__ cbb,
    const _Float16* __restrict__ W1, const float* __restrict__ b1,
    const float* __restrict__ g1, const float* __restrict__ bb1,
    const _Float16* __restrict__ W2, const float* __restrict__ b2,
    const float* __restrict__ g2, const float* __restrict__ bb2,
    float* __restrict__ outbuf) {
  __shared__ _Float16 sA[8][16 * 64];
  __shared__ _Float16 sB[8][16 * 64];
  const int tid  = threadIdx.x;
  const int wave = tid >> 5;
  const int lane = tid & 31;
  const int n15  = lane & 15;
  const int hi   = lane >> 4;
  const int n0   = (blockIdx.x * 8 + wave) * 16;

  _Float16* aT = sA[wave];
  _Float16* bT = sB[wave];

  // stage bn(agg) as f16: lane handles row n15, cols [hi*32, hi*32+32)
  {
    int node = n0 + n15; if (node >= Nn) node = Nn - 1;
    const float* src = agg + (size_t)node * Hdim;
#pragma unroll
    for (int c = hi * 32; c < hi * 32 + 32; c += 4) {
      float4 v = *(const float4*)(src + c);
      v4h hv;
      hv.x = (_Float16)(v.x * (cbg[c + 0] * (float)INVSTD) + cbb[c + 0]);
      hv.y = (_Float16)(v.y * (cbg[c + 1] * (float)INVSTD) + cbb[c + 1]);
      hv.z = (_Float16)(v.z * (cbg[c + 2] * (float)INVSTD) + cbb[c + 2]);
      hv.w = (_Float16)(v.w * (cbg[c + 3] * (float)INVSTD) + cbb[c + 3]);
      *(v4h*)(aT + n15 * 64 + c) = hv;
    }
  }

  // GEMM1 + bn1 + relu -> bT
  v16h a0 = load_frag(aT, 0, 64, 0, lane);
  v16h a1 = load_frag(aT, 0, 64, 32, lane);
#pragma unroll
  for (int q = 0; q < 4; ++q) {
    const int col = q * 16 + n15;
    float bz = b1[col];
    v8f acc;
#pragma unroll
    for (int r = 0; r < 8; ++r) acc[r] = bz;
    v16h w = load_frag(W1, q * 16, 64, 0, lane);
    acc = __builtin_amdgcn_wmma_f32_16x16x32_f16(false, a0, false, w, (short)0, acc, false, false);
    w = load_frag(W1, q * 16, 64, 32, lane);
    acc = __builtin_amdgcn_wmma_f32_16x16x32_f16(false, a1, false, w, (short)0, acc, false, false);
    float sg = g1[col] * (float)INVSTD, sb = bb1[col];
#pragma unroll
    for (int r = 0; r < 8; ++r) {
      float v = fmaxf(acc[r] * sg + sb, 0.f);
      bT[(hi * 8 + r) * 64 + col] = (_Float16)v;
    }
  }

  // GEMM2 + bn2 + relu -> global
  v16h c0 = load_frag(bT, 0, 64, 0, lane);
  v16h c1 = load_frag(bT, 0, 64, 32, lane);
#pragma unroll
  for (int q = 0; q < 4; ++q) {
    const int col = q * 16 + n15;
    float bz = b2[col];
    v8f acc;
#pragma unroll
    for (int r = 0; r < 8; ++r) acc[r] = bz;
    v16h w = load_frag(W2, q * 16, 64, 0, lane);
    acc = __builtin_amdgcn_wmma_f32_16x16x32_f16(false, c0, false, w, (short)0, acc, false, false);
    w = load_frag(W2, q * 16, 64, 32, lane);
    acc = __builtin_amdgcn_wmma_f32_16x16x32_f16(false, c1, false, w, (short)0, acc, false, false);
    float sg = g2[col] * (float)INVSTD, sb = bb2[col];
#pragma unroll
    for (int r = 0; r < 8; ++r) {
      int node = n0 + hi * 8 + r;
      if (node < Nn)
        outbuf[(size_t)node * Hdim + col] = fmaxf(acc[r] * sg + sb, 0.f);
    }
  }
}

// ---------------------------------------------------------------------------
// attention: scores[k,h] = sum_n W0[n,h]*conv_k[n,h] (partial sums + atomics)
// ---------------------------------------------------------------------------
__global__ void attn_scores_kernel(const float* __restrict__ W0,
                                   const float* __restrict__ convr,
                                   float* __restrict__ scores) {
  const int h = threadIdx.x;  // 64 threads
  const float* c0 = convr;
  const float* c1 = convr + (size_t)Nn * Hdim;
  const float* c2 = convr + 2 * (size_t)Nn * Hdim;
  float s0 = 0.f, s1 = 0.f, s2 = 0.f;
  for (int n = blockIdx.x; n < Nn; n += gridDim.x) {
    float w = W0[(size_t)n * Hdim + h];
    s0 += w * c0[(size_t)n * Hdim + h];
    s1 += w * c1[(size_t)n * Hdim + h];
    s2 += w * c2[(size_t)n * Hdim + h];
  }
  atomicAdd(&scores[h], s0);
  atomicAdd(&scores[64 + h], s1);
  atomicAdd(&scores[128 + h], s2);
}

__global__ void attn_final_kernel(const float* __restrict__ scores,
                                  const float* __restrict__ attn_W,
                                  const float* __restrict__ attn_b,
                                  float* __restrict__ wvec) {
  if (threadIdx.x == 0 && blockIdx.x == 0) {
    float logits[3];
    for (int k = 0; k < 3; ++k) {
      float mx = -1e30f, mn = 1e30f;
      for (int h = 0; h < 64; ++h) {
        float v = scores[k * 64 + h];
        mx = fmaxf(mx, v); mn = fminf(mn, v);
      }
      float inv = 1.f / (mx - mn + 1e-6f);
      float acc = 0.f;
      for (int h = 0; h < 64; ++h) acc += (scores[k * 64 + h] - mn) * inv * attn_W[h];
      logits[k] = acc + attn_b[0];
    }
    float m = fmaxf(logits[0], fmaxf(logits[1], logits[2]));
    float e0 = __expf(logits[0] - m), e1 = __expf(logits[1] - m), e2 = __expf(logits[2] - m);
    float inv = 1.f / (e0 + e1 + e2);
    wvec[0] = e0 * inv; wvec[1] = e1 * inv; wvec[2] = e2 * inv;
  }
}

// rep = sum_k w[k]*conv_k + W0 ; pooled[batch[n]] += rep
__global__ void readout_kernel(const float* __restrict__ W0,
                               const float* __restrict__ convr,
                               const float* __restrict__ wvec,
                               const int* __restrict__ batch,
                               float* __restrict__ pooled) {
  int i = blockIdx.x * blockDim.x + threadIdx.x;
  if (i >= Nn * Hdim) return;
  int n = i >> 6, h = i & 63;
  float w0 = wvec[0], w1 = wvec[1], w2 = wvec[2];
  float rep = w0 * convr[i] + w1 * convr[(size_t)Nn * Hdim + i] +
              w2 * convr[2 * (size_t)Nn * Hdim + i] + W0[i];
  atomicAdd(&pooled[(size_t)batch[n] * Hdim + h], rep);
}

// out = relu(pooled@lin1^T+b1)@lin2^T + b2
__global__ void head_kernel(const float* __restrict__ pooled,
                            const float* __restrict__ l1W, const float* __restrict__ l1b,
                            const float* __restrict__ l2W, const float* __restrict__ l2b,
                            float* __restrict__ out) {
  __shared__ float sp[64];
  __shared__ float sh[64];
  const int b = blockIdx.x, t = threadIdx.x;
  sp[t] = pooled[(size_t)b * Hdim + t];
  __syncthreads();
  float a = l1b[t];
  for (int k = 0; k < 64; ++k) a += sp[k] * l1W[t * 64 + k];
  sh[t] = fmaxf(a, 0.f);
  __syncthreads();
  if (t < NCc) {
    float a2 = l2b[t];
    for (int k = 0; k < 64; ++k) a2 += sh[k] * l2W[t * 64 + k];
    out[b * NCc + t] = a2;
  }
}

// ---------------------------------------------------------------------------
extern "C" void kernel_launch(void* const* d_in, const int* in_sizes, int n_in,
                              void* d_out, int out_size, void* d_ws, size_t ws_size,
                              hipStream_t stream) {
  const int* x_ids         = (const int*)d_in[0];
  const int* edge_attr_ids = (const int*)d_in[1];
  const int* paths_[3]     = {(const int*)d_in[2], (const int*)d_in[4], (const int*)d_in[6]};
  const int* eidx_[3]      = {(const int*)d_in[3], (const int*)d_in[5], (const int*)d_in[7]};
  const int* batch         = (const int*)d_in[8];
  const float* feat_emb = (const float*)d_in[9];
  const float* bond_emb = (const float*)d_in[10];
  const float* W_ih = (const float*)d_in[11];
  const float* W_hh = (const float*)d_in[12];
  const float* b_ih = (const float*)d_in[13];
  const float* b_hh = (const float*)d_in[14];
  const float* conv_bn_g = (const float*)d_in[15];
  const float* conv_bn_b = (const float*)d_in[16];
  const float* mlp_W1 = (const float*)d_in[17];
  const float* mlp_b1 = (const float*)d_in[18];
  const float* bn1_g = (const float*)d_in[19];
  const float* bn1_b = (const float*)d_in[20];
  const float* mlp_W2 = (const float*)d_in[21];
  const float* mlp_b2 = (const float*)d_in[22];
  const float* bn2_g = (const float*)d_in[23];
  const float* bn2_b = (const float*)d_in[24];
  const float* attn_W = (const float*)d_in[25];
  const float* attn_b = (const float*)d_in[26];
  const float* lin1_W = (const float*)d_in[27];
  const float* lin1_b = (const float*)d_in[28];
  const float* lin2_W = (const float*)d_in[29];
  const float* lin2_b = (const float*)d_in[30];
  float* out = (float*)d_out;

  const size_t NH = (size_t)Nn * Hdim;
  float* fws    = (float*)d_ws;
  float* W0     = fws;
  float* Win    = W0 + NH;
  float* agg    = Win + NH;
  float* convr  = agg + NH;          // 3*NH
  float* scores = convr + 3 * NH;    // 192
  float* wvec   = scores + 192;      // 16 (padded)
  float* pooled = wvec + 16;         // B*H
  float* bsum   = pooled + (size_t)Bb * Hdim;  // 256
  _Float16* Wih_h = (_Float16*)(bsum + 256);   // 256*128, then Whh contiguous
  _Float16* Whh_h = Wih_h + 256 * 128;
  _Float16* W1_h  = Whh_h + 256 * 64;
  _Float16* W2_h  = W1_h + 3 * 64 * 64;

  hipMemsetAsync(scores, 0, 192 * sizeof(float), stream);
  hipMemsetAsync(pooled, 0, (size_t)Bb * Hdim * sizeof(float), stream);

  prep_kernel<<<128, 256, 0, stream>>>(W_ih, W_hh, mlp_W1, mlp_W2, b_ih, b_hh,
                                       Wih_h, Whh_h, W1_h, W2_h, bsum);
  embed_kernel<<<(Nn * Hdim + 255) / 256, 256, 0, stream>>>(x_ids, feat_emb, W0);

  for (int layer = 0; layer < 3; ++layer) {
    const int L = layer + 2;
    win_kernel<<<(Nn * Hdim + 255) / 256, 256, 0, stream>>>(W0, convr, Win, layer);
    hipMemsetAsync(agg, 0, NH * sizeof(float), stream);
    lstm_kernel<<<(Pp + 127) / 128, 256, 0, stream>>>(
        Win, bond_emb, edge_attr_ids, paths_[layer], eidx_[layer],
        Wih_h, bsum, agg, L, Pp);
    mlp_kernel<<<(Nn + 127) / 128, 256, 0, stream>>>(
        agg, conv_bn_g + layer * 64, conv_bn_b + layer * 64,
        W1_h + layer * 4096, mlp_b1 + layer * 64, bn1_g + layer * 64, bn1_b + layer * 64,
        W2_h + layer * 4096, mlp_b2 + layer * 64, bn2_g + layer * 64, bn2_b + layer * 64,
        convr + (size_t)layer * NH);
  }

  attn_scores_kernel<<<256, 64, 0, stream>>>(W0, convr, scores);
  attn_final_kernel<<<1, 32, 0, stream>>>(scores, attn_W, attn_b, wvec);
  readout_kernel<<<(Nn * Hdim + 255) / 256, 256, 0, stream>>>(W0, convr, wvec, batch, pooled);
  head_kernel<<<Bb, 64, 0, stream>>>(pooled, lin1_W, lin1_b, lin2_W, lin2_b, out);
}
